// RBFN_4105988734985
// MI455X (gfx1250) — compile-verified, matrix operand order checked by reading, not run on hardware
//
#include <hip/hip_runtime.h>
#include <hip/hip_bf16.h>

// ---------------------------------------------------------------------------
// RBF network, fully fused for gfx1250 (MI455X):
//   G   = xs @ centre.T          (K = 128)
//   phi = sqrt(1 + max(x2 + c2 - 2G, 0))
//   out = phi @ weight           (K = 8192, N = 64)
// bf16 WMMA (v_wmma_f32_16x16x32_bf16) with 3-term hi/lo error compensation
// on both GEMMs for ~fp32 accuracy. The 256 MB phi intermediate never leaves
// registers/LDS. All global accesses are uniform-base + 32-bit lane offset so
// the backend can use SADDR addressing (no per-lane 64-bit pointer math).
// ---------------------------------------------------------------------------

typedef __attribute__((ext_vector_type(16))) __bf16 v16bf;
typedef __attribute__((ext_vector_type(8)))  __bf16 v8bf;
typedef __attribute__((ext_vector_type(8)))  float  v8f;

#define D_ROWS 8192
#define N_CENT 8192
#define N_FEAT 128
#define L_OUT  64

// ---------------------------------------------------------------------------
// Fragment helpers. For 16-bit WMMA operands, lane L (h = L>>4) holds
// K = {kb + 8h .. kb + 8h + 7} in VGPRs 0..3 (packed pairs) and
// K = {kb + 16 + 8h .. + 7} in VGPRs 4..7. Both runs are K-contiguous, so a
// fragment is exactly two 16-byte loads from a K-contiguous row.
// ---------------------------------------------------------------------------
__device__ __forceinline__ v16bf frag_cat(v8bf a, v8bf b) {
  return __builtin_shufflevector(a, b, 0, 1, 2, 3, 4, 5, 6, 7,
                                       8, 9, 10, 11, 12, 13, 14, 15);
}

// Global version: uniform base pointer + 32-bit per-lane element offset.
__device__ __forceinline__ v16bf load_frag_g(const __bf16* __restrict__ base,
                                             int off, int h) {
  const v8bf* p0 = reinterpret_cast<const v8bf*>(base + off + 8 * h);
  const v8bf* p1 = reinterpret_cast<const v8bf*>(base + off + 16 + 8 * h);
  return frag_cat(*p0, *p1);
}

// LDS version.
__device__ __forceinline__ v16bf load_frag_lds(const __bf16* rowp, int h) {
  const v8bf* p0 = reinterpret_cast<const v8bf*>(rowp + 8 * h);
  const v8bf* p1 = reinterpret_cast<const v8bf*>(rowp + 16 + 8 * h);
  return frag_cat(*p0, *p1);
}

__device__ __forceinline__ v8f wmma_bf16(v16bf a, v16bf b, v8f c) {
  return __builtin_amdgcn_wmma_f32_16x16x32_bf16(
      /*neg_a=*/false, a, /*neg_b=*/false, b,
      /*c_mod=*/(short)0, c, /*reuse_a=*/false, /*reuse_b=*/false);
}

// ---------------------------------------------------------------------------
// Pre-pass kernels (bandwidth trivial: ~10 MB total)
// ---------------------------------------------------------------------------
__global__ void split_f32_bf16(const float* __restrict__ in,
                               __bf16* __restrict__ hi, __bf16* __restrict__ lo,
                               int count) {
  int i = blockIdx.x * blockDim.x + threadIdx.x;
  if (i < count) {
    float v  = in[i];
    __bf16 h = (__bf16)v;
    hi[i]    = h;
    lo[i]    = (__bf16)(v - (float)h);
  }
}

__global__ void transpose_split(const float* __restrict__ w,   // [N][L]
                                __bf16* __restrict__ hi,       // [L][N]
                                __bf16* __restrict__ lo, int N, int L) {
  int i = blockIdx.x * blockDim.x + threadIdx.x;
  if (i < N * L) {
    int c    = i / L;
    int j    = i - c * L;
    float v  = w[i];
    __bf16 h = (__bf16)v;
    hi[(size_t)j * N + c] = h;
    lo[(size_t)j * N + c] = (__bf16)(v - (float)h);
  }
}

// out[r] = bias + sum(row^2)  (bias 1 folds the RBF's "1 +" into the norm)
__global__ void rowsum2(const float* __restrict__ in, float* __restrict__ out,
                        int rows, int cols, float bias) {
  int r = blockIdx.x * blockDim.x + threadIdx.x;
  if (r < rows) {
    const float* p = in + (size_t)r * cols;
    float s = bias;
    for (int k = 0; k < cols; ++k) s += p[k] * p[k];
    out[r] = s;
  }
}

// ---------------------------------------------------------------------------
// Main fused kernel: 64 blocks x 8 waves; wave owns 16 output rows.
// ---------------------------------------------------------------------------
__global__ __launch_bounds__(256) void rbfn_main(
    const __bf16* __restrict__ xs_hi, const __bf16* __restrict__ xs_lo,
    const __bf16* __restrict__ c_hi,  const __bf16* __restrict__ c_lo,
    const __bf16* __restrict__ wt_hi, const __bf16* __restrict__ wt_lo,
    const float* __restrict__ x2p1,   // 1 + ||x||^2
    const float* __restrict__ c2,     // ||c||^2
    float* __restrict__ out) {
  // Per-wave LDS transpose tile for phi (C-layout -> A-layout), bf16 hi/lo.
  // Row stride 40 elems (80 B): 16-byte aligned rows, 16 distinct LDS banks
  // across the 16 rows read by a ds b128 fragment load.
  constexpr int RS = 40;
  __shared__ alignas(16) __bf16 lds_phi_hi[8][16][RS];
  __shared__ alignas(16) __bf16 lds_phi_lo[8][16][RS];

  const int lane = threadIdx.x & 31;
  const int wave = threadIdx.x >> 5;
  const int h    = lane >> 4;   // K sub-pattern selector
  const int ln   = lane & 15;   // row/col within 16
  const int rowbase = blockIdx.x * 128 + wave * 16;

  // A-matrix fragments of xs (fixed for the whole N loop): 4 K-chunks, hi+lo.
  v16bf a_hi[4], a_lo[4];
  {
    const int xoff = (rowbase + ln) * N_FEAT;
#pragma unroll
    for (int kc = 0; kc < 4; ++kc) {
      a_hi[kc] = load_frag_g(xs_hi, xoff + kc * 32, h);
      a_lo[kc] = load_frag_g(xs_lo, xoff + kc * 32, h);
    }
  }

  // 1 + ||x||^2 for the 8 C-layout rows this lane touches (fp32-exact).
  float x2v[8];
#pragma unroll
  for (int v = 0; v < 8; ++v) x2v[v] = x2p1[rowbase + v + 8 * h];

  v8f acc[4];
#pragma unroll
  for (int nt = 0; nt < 4; ++nt)
    acc[nt] = v8f{0.f, 0.f, 0.f, 0.f, 0.f, 0.f, 0.f, 0.f};

  __bf16* phi_hi = &lds_phi_hi[wave][0][0];
  __bf16* phi_lo = &lds_phi_lo[wave][0][0];

  for (int cb = 0; cb < N_CENT; cb += 32) {
    if (cb + 32 < N_CENT) {  // uniform branch: EXEC stays all-ones for WMMA
      __builtin_prefetch(c_hi + (size_t)((cb + 32 + ln) * N_FEAT), 0, 0);
      __builtin_prefetch(c_hi + (size_t)((cb + 48 + ln) * N_FEAT), 0, 0);
    }

    // ---- GEMM1 + phi for two 16x16 centre tiles ----
#pragma unroll
    for (int ct = 0; ct < 2; ++ct) {
      const int coff = (cb + ct * 16 + ln) * N_FEAT;

      v8f g = v8f{0.f, 0.f, 0.f, 0.f, 0.f, 0.f, 0.f, 0.f};
#pragma unroll
      for (int kc = 0; kc < 4; ++kc) {
        v16bf b_hi = load_frag_g(c_hi, coff + kc * 32, h);
        v16bf b_lo = load_frag_g(c_lo, coff + kc * 32, h);
        g = wmma_bf16(a_hi[kc], b_hi, g);   // hi*hi
        g = wmma_bf16(a_hi[kc], b_lo, g);   // hi*lo
        g = wmma_bf16(a_lo[kc], b_hi, g);   // lo*hi
      }

      const float c2v = c2[cb + ct * 16 + ln];
#pragma unroll
      for (int v = 0; v < 8; ++v) {
        // C-layout: lane -> column, VGPR v -> row (v + 8*h)
        // 1 + max(r2,0) == max((1+x2) + c2 - 2G, 1)
        float t   = fmaf(-2.0f, g[v], x2v[v] + c2v);
        t         = fmaxf(t, 1.0f);
        float phi = __builtin_amdgcn_sqrtf(t);  // arg >= 1: raw v_sqrt_f32 ok
        __bf16 ph = (__bf16)phi;
        __bf16 pl = (__bf16)(phi - (float)ph);
        const int idx = (v + 8 * h) * RS + ct * 16 + ln;
        phi_hi[idx] = ph;
        phi_lo[idx] = pl;
      }
    }

    // Wave-private LDS round trip (no cross-wave sharing -> no barrier).
    asm volatile("s_wait_dscnt 0" ::: "memory");

    // ---- GEMM2: out[16x64] += phi[16x32] @ W[32x64] ----
    v16bf pa_hi = load_frag_lds(phi_hi + ln * RS, h);
    v16bf pa_lo = load_frag_lds(phi_lo + ln * RS, h);

#pragma unroll
    for (int nt = 0; nt < 4; ++nt) {
      const int woff = (nt * 16 + ln) * N_CENT + cb;
      v16bf wb_hi = load_frag_g(wt_hi, woff, h);
      v16bf wb_lo = load_frag_g(wt_lo, woff, h);
      acc[nt] = wmma_bf16(pa_hi, wb_hi, acc[nt]);
      acc[nt] = wmma_bf16(pa_hi, wb_lo, acc[nt]);
      acc[nt] = wmma_bf16(pa_lo, wb_hi, acc[nt]);
    }
    // Next iteration's DS stores cannot pass this iteration's DS loads:
    // LDS ops are in-order within a wave.
  }

  // Epilogue: C-layout -> out[8192][64]
#pragma unroll
  for (int nt = 0; nt < 4; ++nt)
#pragma unroll
    for (int v = 0; v < 8; ++v)
      out[(rowbase + v + 8 * h) * L_OUT + nt * 16 + ln] = acc[nt][v];
}

// ---------------------------------------------------------------------------
// Launch: pre-pass splits/transposes into d_ws (~10.1 MB), then fused kernel.
// ---------------------------------------------------------------------------
extern "C" void kernel_launch(void* const* d_in, const int* in_sizes, int n_in,
                              void* d_out, int out_size, void* d_ws, size_t ws_size,
                              hipStream_t stream) {
  (void)in_sizes; (void)n_in; (void)out_size; (void)ws_size;

  const float* xs     = (const float*)d_in[0];  // [8192][128]
  const float* centre = (const float*)d_in[1];  // [8192][128]
  const float* weight = (const float*)d_in[2];  // [8192][64]
  float* out          = (float*)d_out;          // [8192][64]

  const size_t XS = (size_t)D_ROWS * N_FEAT;    // 1048576
  const size_t WT = (size_t)N_CENT * L_OUT;     // 524288

  __bf16* xs_hi = (__bf16*)d_ws;
  __bf16* xs_lo = xs_hi + XS;
  __bf16* c_hi  = xs_lo + XS;
  __bf16* c_lo  = c_hi + XS;
  __bf16* wt_hi = c_lo + XS;                    // transposed [64][8192]
  __bf16* wt_lo = wt_hi + WT;
  float*  x2p1  = (float*)(wt_lo + WT);         // byte offset 10485760 (4-aligned)
  float*  c2    = x2p1 + D_ROWS;

  const int T = 256;
  split_f32_bf16<<<dim3((XS + T - 1) / T), dim3(T), 0, stream>>>(xs, xs_hi, xs_lo, (int)XS);
  split_f32_bf16<<<dim3((XS + T - 1) / T), dim3(T), 0, stream>>>(centre, c_hi, c_lo, (int)XS);
  transpose_split<<<dim3((WT + T - 1) / T), dim3(T), 0, stream>>>(weight, wt_hi, wt_lo, N_CENT, L_OUT);
  rowsum2<<<dim3((D_ROWS + T - 1) / T), dim3(T), 0, stream>>>(xs, x2p1, D_ROWS, N_FEAT, 1.0f);
  rowsum2<<<dim3((N_CENT + T - 1) / T), dim3(T), 0, stream>>>(centre, c2, N_CENT, N_FEAT, 0.0f);

  rbfn_main<<<dim3(D_ROWS / 128), dim3(T), 0, stream>>>(
      xs_hi, xs_lo, c_hi, c_lo, wt_hi, wt_lo, x2p1, c2, out);
}
